// ContrastivePuzzleLoss_14413910245867
// MI455X (gfx1250) — matrix operand order repaired and checked by reading, not compiled, not run on hardware
//
#include <hip/hip_runtime.h>
#include <math.h>

typedef __attribute__((ext_vector_type(16))) _Float16 v16h;
typedef __attribute__((ext_vector_type(8)))  _Float16 v8h;
typedef __attribute__((ext_vector_type(8)))  float    v8f;

#define BB 128
#define NN 256
#define DD 512
#define KC 64          // K chunk held in LDS
#define NC (DD / KC)   // 8 chunks
#define PITCH 72       // LDS row pitch in halfs (64 + 8 pad; rows are 144B, 16B-aligned)
#define TEMP_INV 14.2857142857142857f   // 1/0.07
#define EPS_F 1e-8f

// ---------------- kernel 0: zero the scalar output ----------------
__global__ void zero_out_kernel(float* out) {
    if (threadIdx.x == 0 && blockIdx.x == 0) out[0] = 0.0f;
}

// helper: scale 8 f32 -> 8 f16
__device__ inline v8h cvt8(const float4 a, const float4 b, float rn) {
    v8h h;
    h[0] = (_Float16)(a.x * rn); h[1] = (_Float16)(a.y * rn);
    h[2] = (_Float16)(a.z * rn); h[3] = (_Float16)(a.w * rn);
    h[4] = (_Float16)(b.x * rn); h[5] = (_Float16)(b.y * rn);
    h[6] = (_Float16)(b.z * rn); h[7] = (_Float16)(b.w * rn);
    return h;
}

// ============ PRIMARY PATH ============
// kernel 1a: L2-normalize rows, convert to f16 once, int64->int32 positions.
__global__ __launch_bounds__(256) void prep_f16_kernel(const float* __restrict__ feats,
                                                       const long long* __restrict__ positions,
                                                       _Float16* __restrict__ f16n,
                                                       int* __restrict__ pos32) {
    int gid  = blockIdx.x * blockDim.x + threadIdx.x;
    int row  = gid >> 5;            // one wave32 per row, 512 elems -> 16 per lane
    int lane = gid & 31;
    const float* src = feats + (size_t)row * DD + lane * 16;
    float4 f0 = *(const float4*)(src + 0);
    float4 f1 = *(const float4*)(src + 4);
    float4 f2 = *(const float4*)(src + 8);
    float4 f3 = *(const float4*)(src + 12);
    float ss = f0.x*f0.x + f0.y*f0.y + f0.z*f0.z + f0.w*f0.w
             + f1.x*f1.x + f1.y*f1.y + f1.z*f1.z + f1.w*f1.w
             + f2.x*f2.x + f2.y*f2.y + f2.z*f2.z + f2.w*f2.w
             + f3.x*f3.x + f3.y*f3.y + f3.z*f3.z + f3.w*f3.w;
#pragma unroll
    for (int off = 16; off >= 1; off >>= 1) ss += __shfl_xor(ss, off, 32);
    float rn = 1.0f / fmaxf(sqrtf(ss), 1e-12f);
    _Float16* dst = f16n + (size_t)row * DD + lane * 16;
    *(v8h*)(dst)     = cvt8(f0, f1, rn);
    *(v8h*)(dst + 8) = cvt8(f2, f3, rn);
    if (lane == 0) pos32[row] = (int)positions[row];
}

// issue the 8 per-thread async b128 transfers that fill one 256x64 f16 panel
__device__ __forceinline__ void fill_panel_async(_Float16* sbuf, const _Float16* fbase,
                                                 int kc, int tid) {
    const int seg = tid & 3;        // 4 threads per row, 32B (2 x b128) each
#pragma unroll
    for (int it = 0; it < 4; ++it) {
        int row = (tid >> 2) + it * 64;
        unsigned lds = (unsigned)(uintptr_t)&sbuf[row * PITCH + seg * 16];
        unsigned long long ga =
            (unsigned long long)(uintptr_t)(fbase + (size_t)row * DD + kc * KC + seg * 16);
        // INST_OFFSET applies to both LDS and memory addresses (ISA 08 §4.4)
        asm volatile("global_load_async_to_lds_b128 %0, %1, off\n\t"
                     "global_load_async_to_lds_b128 %0, %1, off offset:16"
                     :: "v"(lds), "v"(ga) : "memory");
    }
}

// kernel 2a: batched Gram via WMMA; double-buffered async memory->LDS pipeline.
// grid = (16 rowTiles, 128 batches), block = 256 threads (8 waves, 2 colTiles each)
__global__ __launch_bounds__(256) void gram_loss_async_kernel(const _Float16* __restrict__ f16n,
                                                              const int* __restrict__ pos,
                                                              float* __restrict__ out) {
    __shared__ __align__(16) _Float16 sF[2][NN * PITCH];   // 2 x 36 KB f16 K-panels
    __shared__ int   sPos[NN];
    __shared__ float sPosSum[16];
    __shared__ float sTotSum[16];

    const int tid = threadIdx.x;
    const int b   = blockIdx.y;
    const int rt  = blockIdx.x;
    const int rowBase = rt * 16;

    if (tid < 16) { sPosSum[tid] = 0.0f; sTotSum[tid] = 0.0f; }
    sPos[tid] = pos[b * NN + tid];

    const int lane = tid & 31;
    const int wid  = tid >> 5;       // 0..7
    const int half = lane >> 4;      // 0/1
    const int l    = lane & 15;
    const int c0   = wid * 2;
    const int c1   = wid * 2 + 1;

    const _Float16* fbase = f16n + (size_t)b * NN * DD;

    v8f acc0 = {};
    v8f acc1 = {};

    // prologue: start DMA of chunk 0
    fill_panel_async(sF[0], fbase, 0, tid);

    for (int kc = 0; kc < NC; ++kc) {
        const _Float16* cur = sF[kc & 1];
        if (kc + 1 < NC) {
            // start DMA of chunk kc+1 into the other panel (readers finished at
            // the end-of-iteration barrier of kc-1), then wait only for chunk kc:
            // async loads complete in order per wave, 8 outstanding belong to kc+1.
            fill_panel_async((_Float16*)sF[(kc + 1) & 1], fbase, kc + 1, tid);
            asm volatile("s_wait_asynccnt 0x8" ::: "memory");
        } else {
            asm volatile("s_wait_asynccnt 0x0" ::: "memory");
        }
        __syncthreads();             // chunk kc visible to every wave

        // ---- two 16x16x32 WMMA k-steps over this 64-wide chunk ----
#pragma unroll
        for (int ks = 0; ks < 2; ++ks) {
            const int kb = ks * 32;
            // A fragment (ISA 16-bit A 16x32 layout)
            const _Float16* ap = &cur[(rowBase + l) * PITCH + kb + half * 8];
            v8h alo = *(const v8h*)ap;
            v8h ahi = *(const v8h*)(ap + 16);
            v16h a = __builtin_shufflevector(alo, ahi, 0, 1, 2, 3, 4, 5, 6, 7,
                                             8, 9, 10, 11, 12, 13, 14, 15);
            // B fragments (ISA 16-bit B 32x16 layout); B = F^T so B column == F row
            const _Float16* bp0 = &cur[(c0 * 16 + l) * PITCH + kb + half * 8];
            v8h b0lo = *(const v8h*)bp0;
            v8h b0hi = *(const v8h*)(bp0 + 16);
            v16h fb0 = __builtin_shufflevector(b0lo, b0hi, 0, 1, 2, 3, 4, 5, 6, 7,
                                               8, 9, 10, 11, 12, 13, 14, 15);
            const _Float16* bp1 = &cur[(c1 * 16 + l) * PITCH + kb + half * 8];
            v8h b1lo = *(const v8h*)bp1;
            v8h b1hi = *(const v8h*)(bp1 + 16);
            v16h fb1 = __builtin_shufflevector(b1lo, b1hi, 0, 1, 2, 3, 4, 5, 6, 7,
                                               8, 9, 10, 11, 12, 13, 14, 15);

            acc0 = __builtin_amdgcn_wmma_f32_16x16x32_f16(false, a, false, fb0,
                                                          (short)0, acc0, false, false);
            acc1 = __builtin_amdgcn_wmma_f32_16x16x32_f16(false, a, false, fb1,
                                                          (short)0, acc1, false, false);
        }
        __syncthreads();             // all waves done reading cur; it may be refilled
    }

    // ---- masked row reductions straight from the C fragments ----
    // C layout: VGPR r holds (M = r + 8*half, N = l) for this wave's col tiles
    const int mbase = half * 8;
#pragma unroll
    for (int r = 0; r < 8; ++r) {
        const int mloc = r + mbase;
        const int m    = rowBase + mloc;
        const int pm   = sPos[m];
        {
            const int n = c0 * 16 + l;
            float v   = acc0[r];
            float tot = (n == m) ? 0.0f : v;
            float pp  = (n != m && sPos[n] == pm) ? v : 0.0f;
#pragma unroll
            for (int off = 8; off >= 1; off >>= 1) {
                tot += __shfl_xor(tot, off, 32);
                pp  += __shfl_xor(pp,  off, 32);
            }
            if (l == 0) { atomicAdd(&sTotSum[mloc], tot); atomicAdd(&sPosSum[mloc], pp); }
        }
        {
            const int n = c1 * 16 + l;
            float v   = acc1[r];
            float tot = (n == m) ? 0.0f : v;
            float pp  = (n != m && sPos[n] == pm) ? v : 0.0f;
#pragma unroll
            for (int off = 8; off >= 1; off >>= 1) {
                tot += __shfl_xor(tot, off, 32);
                pp  += __shfl_xor(pp,  off, 32);
            }
            if (l == 0) { atomicAdd(&sTotSum[mloc], tot); atomicAdd(&sPosSum[mloc], pp); }
        }
    }
    __syncthreads();

    // ---- 16 rows of this tile: loss = log(exp(pos)+exp(neg)+eps) - pos ----
    if (tid < 16) {
        float ps   = sPosSum[tid] * TEMP_INV;
        float ns   = (sTotSum[tid] - sPosSum[tid]) * TEMP_INV;
        float loss = logf(expf(ps) + expf(ns) + EPS_F) - ps;
        atomicAdd(out, loss * (1.0f / (BB * NN)));
    }
}

// ============ FALLBACK PATH (small workspace): convert f32->f16 in-kernel ============
__global__ __launch_bounds__(256) void prep_kernel(const float* __restrict__ feats,
                                                   const long long* __restrict__ positions,
                                                   float* __restrict__ rnorm,
                                                   int* __restrict__ pos32) {
    int gid  = blockIdx.x * blockDim.x + threadIdx.x;
    int row  = gid >> 5;
    int lane = gid & 31;
    const float* src = feats + (size_t)row * DD + lane * 16;
    float ss = 0.0f;
#pragma unroll
    for (int j = 0; j < 16; j += 4) {
        float4 f = *(const float4*)(src + j);
        ss += f.x * f.x + f.y * f.y + f.z * f.z + f.w * f.w;
    }
#pragma unroll
    for (int off = 16; off >= 1; off >>= 1) ss += __shfl_xor(ss, off, 32);
    if (lane == 0) {
        rnorm[row] = 1.0f / fmaxf(sqrtf(ss), 1e-12f);
        pos32[row] = (int)positions[row];
    }
}

__global__ __launch_bounds__(256) void gram_loss_kernel(const float* __restrict__ feats,
                                                        const float* __restrict__ rnorm,
                                                        const int* __restrict__ pos,
                                                        float* __restrict__ out) {
    __shared__ __align__(16) _Float16 sF[NN * PITCH];
    __shared__ int   sPos[NN];
    __shared__ float sPosSum[16];
    __shared__ float sTotSum[16];

    const int tid = threadIdx.x;
    const int b   = blockIdx.y;
    const int rt  = blockIdx.x;
    const int rowBase = rt * 16;

    if (tid < 16) { sPosSum[tid] = 0.0f; sTotSum[tid] = 0.0f; }
    sPos[tid] = pos[b * NN + tid];

    const int lane = tid & 31;
    const int wid  = tid >> 5;
    const int half = lane >> 4;
    const int l    = lane & 15;
    const int c0   = wid * 2;
    const int c1   = wid * 2 + 1;
    const int seg  = tid & 3;

    const float* fbase = feats + (size_t)b * NN * DD;

    v8f acc0 = {};
    v8f acc1 = {};

    for (int kc = 0; kc < NC; ++kc) {
        __syncthreads();
#pragma unroll
        for (int it = 0; it < 4; ++it) {
            int row = (tid >> 2) + it * 64;
            const float* src = fbase + (size_t)row * DD + kc * KC + seg * 16;
            float rn = rnorm[b * NN + row];
            float4 f0 = *(const float4*)(src + 0);
            float4 f1 = *(const float4*)(src + 4);
            float4 f2 = *(const float4*)(src + 8);
            float4 f3 = *(const float4*)(src + 12);
            _Float16* dst = &sF[row * PITCH + seg * 16];
            *(v8h*)(dst)     = cvt8(f0, f1, rn);
            *(v8h*)(dst + 8) = cvt8(f2, f3, rn);
        }
        __syncthreads();

        if (kc + 1 < NC)
            __builtin_prefetch(fbase + (size_t)tid * DD + (kc + 1) * KC, 0, 1);

#pragma unroll
        for (int ks = 0; ks < 2; ++ks) {
            const int kb = ks * 32;
            const _Float16* ap = &sF[(rowBase + l) * PITCH + kb + half * 8];
            v8h alo = *(const v8h*)ap;
            v8h ahi = *(const v8h*)(ap + 16);
            v16h a = __builtin_shufflevector(alo, ahi, 0, 1, 2, 3, 4, 5, 6, 7,
                                             8, 9, 10, 11, 12, 13, 14, 15);
            const _Float16* bp0 = &sF[(c0 * 16 + l) * PITCH + kb + half * 8];
            v8h b0lo = *(const v8h*)bp0;
            v8h b0hi = *(const v8h*)(bp0 + 16);
            v16h fb0 = __builtin_shufflevector(b0lo, b0hi, 0, 1, 2, 3, 4, 5, 6, 7,
                                               8, 9, 10, 11, 12, 13, 14, 15);
            const _Float16* bp1 = &sF[(c1 * 16 + l) * PITCH + kb + half * 8];
            v8h b1lo = *(const v8h*)bp1;
            v8h b1hi = *(const v8h*)(bp1 + 16);
            v16h fb1 = __builtin_shufflevector(b1lo, b1hi, 0, 1, 2, 3, 4, 5, 6, 7,
                                               8, 9, 10, 11, 12, 13, 14, 15);

            acc0 = __builtin_amdgcn_wmma_f32_16x16x32_f16(false, a, false, fb0,
                                                          (short)0, acc0, false, false);
            acc1 = __builtin_amdgcn_wmma_f32_16x16x32_f16(false, a, false, fb1,
                                                          (short)0, acc1, false, false);
        }
    }

    const int mbase = half * 8;
#pragma unroll
    for (int r = 0; r < 8; ++r) {
        const int mloc = r + mbase;
        const int m    = rowBase + mloc;
        const int pm   = sPos[m];
        {
            const int n = c0 * 16 + l;
            float v   = acc0[r];
            float tot = (n == m) ? 0.0f : v;
            float pp  = (n != m && sPos[n] == pm) ? v : 0.0f;
#pragma unroll
            for (int off = 8; off >= 1; off >>= 1) {
                tot += __shfl_xor(tot, off, 32);
                pp  += __shfl_xor(pp,  off, 32);
            }
            if (l == 0) { atomicAdd(&sTotSum[mloc], tot); atomicAdd(&sPosSum[mloc], pp); }
        }
        {
            const int n = c1 * 16 + l;
            float v   = acc1[r];
            float tot = (n == m) ? 0.0f : v;
            float pp  = (n != m && sPos[n] == pm) ? v : 0.0f;
#pragma unroll
            for (int off = 8; off >= 1; off >>= 1) {
                tot += __shfl_xor(tot, off, 32);
                pp  += __shfl_xor(pp,  off, 32);
            }
            if (l == 0) { atomicAdd(&sTotSum[mloc], tot); atomicAdd(&sPosSum[mloc], pp); }
        }
    }
    __syncthreads();

    if (tid < 16) {
        float ps   = sPosSum[tid] * TEMP_INV;
        float ns   = (sTotSum[tid] - sPosSum[tid]) * TEMP_INV;
        float loss = logf(expf(ps) + expf(ns) + EPS_F) - ps;
        atomicAdd(out, loss * (1.0f / (BB * NN)));
    }
}

extern "C" void kernel_launch(void* const* d_in, const int* in_sizes, int n_in,
                              void* d_out, int out_size, void* d_ws, size_t ws_size,
                              hipStream_t stream) {
    const float*     feats     = (const float*)d_in[0];
    const long long* positions = (const long long*)d_in[1];
    float* out = (float*)d_out;

    const size_t f16_bytes = (size_t)BB * NN * DD * sizeof(_Float16);   // 32 MB
    const size_t idx_bytes = (size_t)BB * NN * sizeof(int);             // 128 KB

    hipLaunchKernelGGL(zero_out_kernel, dim3(1), dim3(64), 0, stream, out);

    if (ws_size >= f16_bytes + idx_bytes) {
        // primary: pre-normalized f16 matrix + double-buffered async LDS pipeline
        _Float16* f16n = (_Float16*)d_ws;
        int* pos32     = (int*)((char*)d_ws + f16_bytes);
        hipLaunchKernelGGL(prep_f16_kernel, dim3((BB * NN * 32) / 256), dim3(256), 0, stream,
                           feats, positions, f16n, pos32);
        hipLaunchKernelGGL(gram_loss_async_kernel, dim3(16, BB), dim3(256), 0, stream,
                           f16n, pos32, out);
    } else {
        // fallback: small workspace, convert f32->f16 during the LDS stage
        float* rnorm = (float*)d_ws;
        int*   pos32 = (int*)((char*)d_ws + (size_t)BB * NN * sizeof(float));
        hipLaunchKernelGGL(prep_kernel, dim3((BB * NN * 32) / 256), dim3(256), 0, stream,
                           feats, positions, rnorm, pos32);
        hipLaunchKernelGGL(gram_loss_kernel, dim3(16, BB), dim3(256), 0, stream,
                           feats, rnorm, pos32, out);
    }
}
